// NeuralDecisionTree_18700287607531
// MI455X (gfx1250) — matrix operand order repaired and checked by reading, not compiled, not run on hardware
//
#include <hip/hip_runtime.h>

typedef _Float16 half_t;
typedef __attribute__((ext_vector_type(16))) _Float16 v16h;
typedef __attribute__((ext_vector_type(2)))  _Float16 half2_t;
typedef __attribute__((ext_vector_type(8)))  float    v8f;

#define DEPTH   6
#define NLEAF   64
#define NINT    63
#define INDIM   512
#define H1DIM   128
#define H2DIM   64
#define ODIM    8
#define BATCH   8192
#define WAVES   8                       // 256 threads per block
#define ROWS_PER_BLOCK (WAVES * 16)     // 128 rows -> 64 row-blocks
#define NGROUPS 8                       // leaf-split factor
#define LPG     (NLEAF / NGROUPS)       // 8 leaves per block

// ---------------- WMMA helpers ----------------

__device__ __forceinline__ v8f wmma_f16(v16h a, v16h b, v8f c) {
    // D = A(16x32 f16) x B(32x16 f16) + C(16x16 f32)
    return __builtin_amdgcn_wmma_f32_16x16x32_f16(
        /*neg_a=*/false, a, /*neg_b=*/false, b,
        /*c_mod=*/(short)0, c, /*reuse_a=*/false, /*reuse_b=*/false);
}

// A fragment (16-bit A 16x32, ISA 7.12.2): lane holds row M = lane&15.
// VGPR v, half h: K = 16*(v>=4) + 8*(lane>=16) + 2*(v&3) + h  (two 16B runs).
__device__ __forceinline__ v16h loadA(const half_t* __restrict__ rowptr, int kbase, int lhi) {
    v16h a;
#pragma unroll
    for (int v = 0; v < 8; ++v) {
        int k = kbase + ((v & 4) << 2) + (lhi << 3) + ((v & 3) << 1);
        half2_t p = *(const half2_t*)(rowptr + k);
        a[2 * v]     = p[0];
        a[2 * v + 1] = p[1];
    }
    return a;
}

// B fragment (16-bit B 32x16): weights stored transposed as [N][K]; lane holds
// column N = lane&15. VGPR v, half h: K = 16*(v>=4) + 8*(lane>=16) + 2*(v&3) + h (two 16B runs).
__device__ __forceinline__ v16h loadB(const half_t* __restrict__ rowptr, int kbase, int lhi) {
    v16h b;
#pragma unroll
    for (int v = 0; v < 8; ++v) {
        int k = kbase + ((v & 4) << 2) + (lhi << 3) + ((v & 3) << 1);
        half2_t p = *(const half2_t*)(rowptr + k);
        b[2 * v]     = p[0];
        b[2 * v + 1] = p[1];
    }
    return b;
}

// ---------------- prep kernels: f32 -> f16, weight transposes ----------------

__global__ void k_cvt_x(const float* __restrict__ x, half_t* __restrict__ xh) {
    int i = blockIdx.x * blockDim.x + threadIdx.x;          // 8192*512
    xh[i] = (half_t)x[i];
}

__global__ void k_pad_rw(const float* __restrict__ rw, half_t* __restrict__ rwh) {
    int i = blockIdx.x * blockDim.x + threadIdx.x;          // 64*512 (row 63 zero pad)
    int n = i >> 9, k = i & 511;
    rwh[i] = (n < NINT) ? (half_t)rw[n * INDIM + k] : (half_t)0.0f;
}

__global__ void k_t_w1(const float* __restrict__ w, half_t* __restrict__ wt) {
    int i = blockIdx.x * blockDim.x + threadIdx.x;          // 64*128*512, wt[l][h][k]
    int l = i >> 16, h = (i >> 9) & 127, k = i & 511;
    wt[i] = (half_t)w[((size_t)l * INDIM + k) * H1DIM + h];
}

__global__ void k_t_w2(const float* __restrict__ w, half_t* __restrict__ wt) {
    int i = blockIdx.x * blockDim.x + threadIdx.x;          // 64*64*128, wt[l][h2][h1]
    int l = i >> 13, h2 = (i >> 7) & 63, k = i & 127;
    wt[i] = (half_t)w[((size_t)l * H1DIM + k) * H2DIM + h2];
}

__global__ void k_t_w3(const float* __restrict__ w, half_t* __restrict__ wt) {
    int i = blockIdx.x * blockDim.x + threadIdx.x;          // 64*16*64, wt[l][o(pad16)][k]
    int l = i >> 10, o = (i >> 6) & 15, k = i & 63;
    wt[i] = (o < ODIM) ? (half_t)w[((size_t)l * H2DIM + k) * ODIM + o] : (half_t)0.0f;
}

// deterministic cross-group reduction: out[b,o] = sum_g partial[g][b][o]
__global__ void k_reduce(const float* __restrict__ partial, float* __restrict__ out) {
    int i = blockIdx.x * blockDim.x + threadIdx.x;          // 8192*8
    float s = 0.0f;
#pragma unroll
    for (int g = 0; g < NGROUPS; ++g)
        s += partial[(size_t)g * BATCH * ODIM + i];
    out[i] = s;
}

// ---------------- main fused tree kernel ----------------

__global__ __launch_bounds__(256) void tree_main(
    const half_t* __restrict__ xh,
    const half_t* __restrict__ rWh,
    const float*  __restrict__ rb,
    const half_t* __restrict__ W1h,
    const float*  __restrict__ b1,
    const half_t* __restrict__ W2h,
    const float*  __restrict__ b2,
    const half_t* __restrict__ W3h,
    const float*  __restrict__ b3,
    float*        __restrict__ partial)   // [NGROUPS][BATCH][ODIM]
{
    __shared__ float  p_lds[WAVES][16][LPG];   // leaf probabilities       4 KB
    __shared__ half_t h2_lds[WAVES][16][64];   // layer-2 activations     16 KB
    __shared__ float  ruh[WAVES][1024];        // route(f32)/h1(f16) union 32 KB

    const int tid  = threadIdx.x;
    const int wave = tid >> 5;
    const int lane = tid & 31;
    const int lhi  = lane >> 4;      // half-wave select
    const int l15  = lane & 15;      // M (A) / N (B,C) lane coordinate
    const int rowbase = blockIdx.x * ROWS_PER_BLOCK + wave * 16;
    const int leaf0   = blockIdx.y * LPG;

    float*  route_w = (float*)ruh[wave];   // [16][64] f32  (phase A/B)
    half_t* h1_w    = (half_t*)ruh[wave];  // [16][128] f16 (leaf loop)

    // Preload ALL A-fragments of this wave's 16 x-rows (K=512 -> 16 steps).
    // Reused by the router and by every leaf's layer-1 GEMM (68 uses each).
    v16h ax[16];
    const half_t* xrow = xh + (size_t)(rowbase + l15) * INDIM;
#pragma unroll
    for (int s = 0; s < 16; ++s) ax[s] = loadA(xrow, s * 32, lhi);

    // ---- Phase A: router logits (16 x 64 per wave) -> sigmoid -> LDS ----
#pragma unroll
    for (int nt = 0; nt < 4; ++nt) {
        v8f acc = {};
        const half_t* wrow = rWh + (size_t)(nt * 16 + l15) * INDIM;
        v16h bc = loadB(wrow, 0, lhi);                 // software pipeline
#pragma unroll
        for (int s = 0; s < 15; ++s) {
            v16h bn = loadB(wrow, (s + 1) * 32, lhi);
            acc = wmma_f16(ax[s], bc, acc);
            bc = bn;
        }
        acc = wmma_f16(ax[15], bc, acc);

        int node = nt * 16 + l15;
        float bias = (node < NINT) ? rb[node] : 0.0f;
#pragma unroll
        for (int r = 0; r < 8; ++r) {
            float t = acc[r] + bias;
            route_w[(r + lhi * 8) * 64 + node] = 1.0f / (1.0f + __expf(-t));
        }
    }
    __syncthreads();

    // ---- Phase B: path probabilities for this block's 8 leaves ----
    for (int t = lane; t < 16 * LPG; t += 32) {
        int row = t >> 3;
        int li  = t & (LPG - 1);
        int l   = leaf0 + li;
        float p = 1.0f;
#pragma unroll
        for (int k = 0; k < DEPTH; ++k) {
            int node = (1 << k) - 1 + (l >> (DEPTH - k));
            float d  = route_w[row * 64 + node];
            int bit  = (l >> (DEPTH - 1 - k)) & 1;
            p *= bit ? (1.0f - d) : d;
        }
        p_lds[wave][row][li] = p;
    }
    __syncthreads();   // route_w dead -> h1_w may overwrite it

    v8f outacc = {};

    // ---- Phase C: this block's LPG leaf experts ----
    for (int li = 0; li < LPG; ++li) {
        const int l = leaf0 + li;
        if (li + 1 < LPG)   // warm caches for next leaf's layer-1 weights
            __builtin_prefetch(W1h + (size_t)(l + 1) * H1DIM * INDIM, 0, 0);

        // Layer 1: h1 = relu(x @ W1[l] + b1[l])   (16 x 128, K=512)
#pragma unroll
        for (int nt = 0; nt < 8; ++nt) {
            __syncthreads();  // keep 8 waves in lockstep so WGP$ dedupes W1 reads
            v8f acc = {};
            const half_t* wrow = W1h + (size_t)(l * H1DIM + nt * 16 + l15) * INDIM;
            v16h bc = loadB(wrow, 0, lhi);
#pragma unroll
            for (int s = 0; s < 15; ++s) {
                v16h bn = loadB(wrow, (s + 1) * 32, lhi);
                acc = wmma_f16(ax[s], bc, acc);
                bc = bn;
            }
            acc = wmma_f16(ax[15], bc, acc);

            float bias = b1[l * H1DIM + nt * 16 + l15];
#pragma unroll
            for (int r = 0; r < 8; ++r) {
                float t = acc[r] + bias;
                t = t > 0.0f ? t : 0.0f;
                h1_w[(r + lhi * 8) * H1DIM + nt * 16 + l15] = (half_t)t;  // C->A relayout
            }
        }
        __syncthreads();

        // Layer 2: h2 = relu(h1 @ W2[l] + b2[l])  (16 x 64, K=128)
        v16h a2[4];
#pragma unroll
        for (int s = 0; s < 4; ++s) a2[s] = loadA(h1_w + l15 * H1DIM, s * 32, lhi);
#pragma unroll
        for (int nt = 0; nt < 4; ++nt) {
            v8f acc = {};
            const half_t* wrow = W2h + (size_t)(l * H2DIM + nt * 16 + l15) * H1DIM;
            v16h bc = loadB(wrow, 0, lhi);
#pragma unroll
            for (int s = 0; s < 3; ++s) {
                v16h bn = loadB(wrow, (s + 1) * 32, lhi);
                acc = wmma_f16(a2[s], bc, acc);
                bc = bn;
            }
            acc = wmma_f16(a2[3], bc, acc);

            float bias = b2[l * H2DIM + nt * 16 + l15];
#pragma unroll
            for (int r = 0; r < 8; ++r) {
                float t = acc[r] + bias;
                t = t > 0.0f ? t : 0.0f;
                h2_lds[wave][r + lhi * 8][nt * 16 + l15] = (half_t)t;
            }
        }
        __syncthreads();

        // Layer 3: pred = h2 @ W3[l] + b3[l]      (16 x 8 in a padded 16x16, K=64)
        v16h a3[2];
#pragma unroll
        for (int s = 0; s < 2; ++s) a3[s] = loadA(&h2_lds[wave][l15][0], s * 32, lhi);
        v8f accp = {};
        {
            const half_t* wrow = W3h + (size_t)(l * 16 + l15) * H2DIM;
            v16h b0 = loadB(wrow, 0, lhi);
            v16h b1f = loadB(wrow, 32, lhi);
            accp = wmma_f16(a3[0], b0, accp);
            accp = wmma_f16(a3[1], b1f, accp);
        }
        float b3v = (l15 < ODIM) ? b3[l * ODIM + l15] : 0.0f;
#pragma unroll
        for (int r = 0; r < 8; ++r) {
            float pv = p_lds[wave][r + lhi * 8][li];
            outacc[r] += pv * (accp[r] + b3v);
        }
        __syncthreads();   // protect h1/h2 LDS reuse next leaf
    }

    // ---- Store this group's partial sums (each (g,row,o) written once) ----
    float* pout = partial + (size_t)blockIdx.y * BATCH * ODIM;
    if (l15 < ODIM) {
#pragma unroll
        for (int r = 0; r < 8; ++r)
            pout[(size_t)(rowbase + r + lhi * 8) * ODIM + l15] = outacc[r];
    }
}

// ---------------- host launcher ----------------

extern "C" void kernel_launch(void* const* d_in, const int* in_sizes, int n_in,
                              void* d_out, int out_size, void* d_ws, size_t ws_size,
                              hipStream_t stream) {
    const float* x  = (const float*)d_in[0];   // 8192 x 512
    const float* rW = (const float*)d_in[1];   // 63 x 512
    const float* rb = (const float*)d_in[2];   // 63
    const float* W1 = (const float*)d_in[3];   // 64 x 512 x 128
    const float* b1 = (const float*)d_in[4];   // 64 x 128
    const float* W2 = (const float*)d_in[5];   // 64 x 128 x 64
    const float* b2 = (const float*)d_in[6];   // 64 x 64
    const float* W3 = (const float*)d_in[7];   // 64 x 64 x 8
    const float* b3 = (const float*)d_in[8];   // 64 x 8
    float* out = (float*)d_out;                // 8192 x 8

    char* ws = (char*)d_ws;
    half_t* xh   = (half_t*)(ws + 0);          //  8 MB : x in f16
    half_t* W1h  = (half_t*)(ws + 8388608);    //  8 MB : W1^T f16 [l][h][k]
    half_t* W2h  = (half_t*)(ws + 16777216);   //  1 MB : W2^T f16 [l][h2][h1]
    half_t* W3h  = (half_t*)(ws + 17825792);   // 128 KB: W3^T f16 [l][o pad16][k]
    half_t* rWh  = (half_t*)(ws + 17956864);   //  64 KB: router_W f16 padded
    float*  part = (float*)(ws + 18022400);    //   2 MB: partial[8][8192][8]

    k_cvt_x <<<(BATCH * INDIM) / 256,          256, 0, stream>>>(x,  xh);
    k_t_w1  <<<(NLEAF * H1DIM * INDIM) / 256,  256, 0, stream>>>(W1, W1h);
    k_t_w2  <<<(NLEAF * H2DIM * H1DIM) / 256,  256, 0, stream>>>(W2, W2h);
    k_t_w3  <<<(NLEAF * 16 * H2DIM) / 256,     256, 0, stream>>>(W3, W3h);
    k_pad_rw<<<(NLEAF * INDIM) / 256,          256, 0, stream>>>(rW, rWh);

    dim3 grid(BATCH / ROWS_PER_BLOCK, NGROUPS);   // 64 x 8 = 512 blocks, 4096 waves
    tree_main<<<grid, 256, 0, stream>>>(
        xh, rWh, rb, W1h, b1, W2h, b2, W3h, b3, part);

    k_reduce<<<(BATCH * ODIM) / 256, 256, 0, stream>>>(part, out);

    (void)in_sizes; (void)n_in; (void)out_size; (void)ws_size;
}
